// WanAttention_18245021073721
// MI455X (gfx1250) — matrix operand order repaired and checked by reading, not compiled, not run on hardware
//
#include <hip/hip_runtime.h>
#include <hip/hip_bf16.h>

// ---------------- problem constants ----------------
#define TT 4096
#define CC 3072
#define HH 24
#define DD 128

typedef __bf16 bf16_t;
typedef __attribute__((ext_vector_type(4)))  bf16_t   v4bf;
typedef __attribute__((ext_vector_type(8)))  bf16_t   v8bf;
typedef __attribute__((ext_vector_type(16))) bf16_t   v16bf;
typedef __attribute__((ext_vector_type(8)))  float    v8f;
typedef __attribute__((ext_vector_type(4)))  unsigned v4u;
typedef __attribute__((ext_vector_type(8)))  int      v8i;
typedef __attribute__((ext_vector_type(4)))  int      v4i;

#if __has_builtin(__builtin_amdgcn_tensor_load_to_lds)
#define USE_TDM 1
#else
#define USE_TDM 0
#endif

// gfx1250 has native bf16: plain casts lower to v_cvt_* when available and
// can never fail to compile (fallback is the generic RNE expansion).
static __device__ __forceinline__ bf16_t f32_to_bf16(float f) { return (bf16_t)f; }
static __device__ __forceinline__ float  bf16_to_f32(bf16_t b){ return (float)b; }

static __device__ __forceinline__ v16bf cat16(v8bf lo, v8bf hi) {
    return __builtin_shufflevector(lo, hi, 0,1,2,3,4,5,6,7,8,9,10,11,12,13,14,15);
}
static __device__ __forceinline__ v8f v8f_zero() {
    v8f z = {0.f,0.f,0.f,0.f,0.f,0.f,0.f,0.f}; return z;
}
// A-fragment (16x32 bf16): per-lane K-run of 16 at k0=16*(lane/16)
static __device__ __forceinline__ v16bf load_a_frag(const bf16_t* p) {
    return cat16(*(const v8bf*)p, *(const v8bf*)(p + 8));
}
// B-fragment (32x16 bf16): per-lane K-run of 16 at k0=16*(lane/16)
static __device__ __forceinline__ v16bf load_b_frag(const bf16_t* p) {
    return cat16(*(const v8bf*)p, *(const v8bf*)(p + 8));
}
static __device__ __forceinline__ void wait_tensorcnt0() {
#if __has_builtin(__builtin_amdgcn_s_wait_tensorcnt)
    __builtin_amdgcn_s_wait_tensorcnt(0);
#else
    asm volatile("s_wait_tensorcnt 0x0" ::: "memory");
#endif
}

// =====================================================================
// GEMM: Out[m,n] = sum_k A[m,k] * W[n,k]   (A: MxK fp32, W: NxK fp32)
// Block tile 128x128, K-step 32, 8 waves -> each wave 64x32 (2x4 grid).
// Double-buffered LDS (one barrier per K-step); fp32 tiles staged with
// on-the-fly bf16 conversion; next-tile prefetch via global_prefetch_b8.
// =====================================================================
template<bool OUT_F32>
__global__ __launch_bounds__(256) void gemm_xw(const float* __restrict__ A,
                                               const float* __restrict__ W,
                                               void* __restrict__ Out,
                                               int M, int N, int K)
{
    __shared__ __align__(16) bf16_t ldsA[2][128 * 40];   // row stride 40 (80B)
    __shared__ __align__(16) bf16_t ldsB[2][128 * 40];

    const int tid  = threadIdx.x;
    const int lane = tid & 31;
    const int wv   = tid >> 5;          // 0..7
    const int wm   = wv & 1;            // 0..1 -> 64 rows each
    const int wn   = wv >> 1;           // 0..3 -> 32 cols each
    const int hh   = lane >> 4;         // 0/1  (lane half)
    const int ll   = lane & 15;

    const int m0 = blockIdx.y * 128;
    const int n0 = blockIdx.x * 128;

    const int lrow = tid >> 1;          // 0..127 (staging row)
    const int lk   = (tid & 1) * 16;    // 0 or 16 (staging k-offset)

    const float* gA = A + (size_t)(m0 + lrow) * K + lk;
    const float* gB = W + (size_t)(n0 + lrow) * K + lk;

    // stage one 128x32 fp32 tile pair -> bf16 LDS buffers
    auto stage = [&](int kt, int buf) {
        const float* ga = gA + kt;
        const float* gb = gB + kt;
        __builtin_prefetch(ga + 32, 0, 0);
        __builtin_prefetch(gb + 32, 0, 0);
#pragma unroll
        for (int i = 0; i < 4; ++i) {
            float4 fa = *(const float4*)(ga + 4 * i);
            float4 fb = *(const float4*)(gb + 4 * i);
            v4bf pa, pb;
            pa[0] = f32_to_bf16(fa.x); pa[1] = f32_to_bf16(fa.y);
            pa[2] = f32_to_bf16(fa.z); pa[3] = f32_to_bf16(fa.w);
            pb[0] = f32_to_bf16(fb.x); pb[1] = f32_to_bf16(fb.y);
            pb[2] = f32_to_bf16(fb.z); pb[3] = f32_to_bf16(fb.w);
            *(v4bf*)(&ldsA[buf][lrow * 40 + lk + 4 * i]) = pa;
            *(v4bf*)(&ldsB[buf][lrow * 40 + lk + 4 * i]) = pb;
        }
    };

    v8f acc[4][2];
#pragma unroll
    for (int mt = 0; mt < 4; ++mt)
#pragma unroll
        for (int nt = 0; nt < 2; ++nt) acc[mt][nt] = v8f_zero();

    stage(0, 0);
    int cur = 0;
    for (int kt = 0; kt < K; kt += 32) {
        __syncthreads();                        // staging of `cur` complete
        if (kt + 32 < K) stage(kt + 32, cur ^ 1);   // overlaps with compute below

        v16bf afr[4], bfr[2];
#pragma unroll
        for (int mt = 0; mt < 4; ++mt)
            afr[mt] = load_a_frag(&ldsA[cur][(wm * 64 + mt * 16 + ll) * 40 + 16 * hh]);
#pragma unroll
        for (int nt = 0; nt < 2; ++nt)
            bfr[nt] = load_b_frag(&ldsB[cur][(wn * 32 + nt * 16 + ll) * 40 + 16 * hh]);

#pragma unroll
        for (int mt = 0; mt < 4; ++mt)
#pragma unroll
            for (int nt = 0; nt < 2; ++nt)
                acc[mt][nt] = __builtin_amdgcn_wmma_f32_16x16x32_bf16(
                    false, afr[mt], false, bfr[nt], (short)0, acc[mt][nt], false, false);
        cur ^= 1;
    }

    // epilogue: C/D layout row = r + 8*(lane/16), col = lane%16
#pragma unroll
    for (int mt = 0; mt < 4; ++mt)
#pragma unroll
        for (int nt = 0; nt < 2; ++nt)
#pragma unroll
            for (int r = 0; r < 8; ++r) {
                const int row = m0 + wm * 64 + mt * 16 + r + 8 * hh;
                const int col = n0 + wn * 32 + nt * 16 + ll;
                const float v = acc[mt][nt][r];
                if (OUT_F32)
                    ((float*)Out)[(size_t)row * N + col] = v;
                else
                    ((bf16_t*)Out)[(size_t)row * N + col] = f32_to_bf16(v);
            }
}

// =====================================================================
// RoPE (in-place on bf16 (T,H,D) buffer). freqs: (T, D/2, 2, 2) fp32.
// =====================================================================
__global__ __launch_bounds__(256) void rope_kernel(bf16_t* __restrict__ X,
                                                   const float* __restrict__ F)
{
    const int idx = blockIdx.x * blockDim.x + threadIdx.x;
    const int total = TT * HH * (DD / 2);
    if (idx >= total) return;
    const int t   = idx / (HH * (DD / 2));
    const int rem = idx % (HH * (DD / 2));
    const int h   = rem / (DD / 2);
    const int i   = rem % (DD / 2);
    const size_t base = (size_t)t * CC + h * DD + 2 * i;
    const float q0 = bf16_to_f32(X[base]);
    const float q1 = bf16_to_f32(X[base + 1]);
    const float* f = F + (size_t)t * (DD * 2) + i * 4;  // [cos, -sin, sin, cos]
    X[base]     = f32_to_bf16(f[0] * q0 + f[1] * q1);
    X[base + 1] = f32_to_bf16(f[2] * q0 + f[3] * q1);
}

// =====================================================================
// Flash attention. One block = 128 query rows of one head; 8 waves own
// 16 rows each; Bc = 64 keys/iteration.
//   K tile  -> staged to LDS by the Tensor Data Mover (TDM) with D#
//              padding (272B row stride) for bank-conflict-free B-frags.
//   V tile  -> staged transposed in LDS by the threads (TDM can't
//              transpose) for the PV B-frags.
//   P       -> bounced through wave-private LDS (C-layout -> A-layout).
// Online softmax via shfl reductions over the 16-lane halves.
// =====================================================================
__global__ __launch_bounds__(256) void flash_attn(const bf16_t* __restrict__ Q,
                                                  const bf16_t* __restrict__ K,
                                                  const bf16_t* __restrict__ V,
                                                  float* __restrict__ O)
{
    __shared__ __align__(16) bf16_t Vt[128 * 72];          // V^T tile: [d][j]
    __shared__ __align__(16) bf16_t Pl[8 * 16 * 72];       // per-wave P: [m][j]
#if USE_TDM
    __shared__ __align__(16) bf16_t Kl[64 * 136];          // K tile: [j][d], 272B stride
#endif

    const int tid  = threadIdx.x;
    const int lane = tid & 31;
    const int wv   = tid >> 5;
    const int hh   = lane >> 4;
    const int ll   = lane & 15;
    const int t0   = blockIdx.x * 128;
    const int h    = blockIdx.y;

    // Q fragments (held for whole kernel): 4 K-steps of 32 over D=128
    v16bf qf[4];
    {
        const bf16_t* qp = Q + (size_t)(t0 + wv * 16 + ll) * CC + h * DD;
#pragma unroll
        for (int ks = 0; ks < 4; ++ks)
            qf[ks] = load_a_frag(qp + ks * 32 + 16 * hh);
    }

    v8f o[8];
#pragma unroll
    for (int nt = 0; nt < 8; ++nt) o[nt] = v8f_zero();
    float mi[8], li[8];
#pragma unroll
    for (int r = 0; r < 8; ++r) { mi[r] = -1e30f; li[r] = 0.f; }

    const float scale = 0.08838834764831845f;   // 1/sqrt(128)

    for (int j0 = 0; j0 < TT; j0 += 64) {
#if USE_TDM
        // ---- TDM: async-stage K tile (64 rows x 128 d, 2B elems) into LDS.
        // D#: data_size=2B, tile 128x64, tensor_dim0_stride=3072,
        //     pad_enable, pad_interval=64 DW (every 256B row), pad_amount=4 DW.
        if (wv == 0) {
            const unsigned long long ga =
                (unsigned long long)(const void*)(K + (size_t)j0 * CC + h * DD);
            const unsigned lds_off = (unsigned)(size_t)(void*)&Kl[0];
            v4u g0 = { 1u,                                  // count=1 valid descriptor
                       lds_off,                             // lds_addr
                       (unsigned)ga,                        // global_addr[31:0]
                       (unsigned)((ga >> 32) & 0x01ffffffu) | (2u << 30) }; // addr[56:32] | type=2
            v8i g1 = { (int)0x07510000,  // data_size=2B, pad_en, pad_interval=5, pad_amount=3
                       (int)(128 << 16), // tensor_dim0 = 128 (low 16 at [63:48])
                       (int)(64  << 16), // tensor_dim0 hi=0 | tensor_dim1 = 64
                       (int)(128 << 16), // tensor_dim1 hi=0 | tile_dim0 = 128
                       64,               // tile_dim1 = 64, tile_dim2 = 0
                       3072,             // tensor_dim0_stride = 3072
                       0, 0 };           // stride hi / tensor_dim1_stride = 0
            v4i gz = { 0, 0, 0, 0 };
#if __clang_major__ >= 23
            v8i gz8 = { 0, 0, 0, 0, 0, 0, 0, 0 };
            __builtin_amdgcn_tensor_load_to_lds(g0, g1, gz, gz, gz8, 0);
#else
            __builtin_amdgcn_tensor_load_to_lds(g0, g1, gz, gz, 0);
#endif
        }
#endif
        // ---- stage V^T tile (64 keys x 128 dims), cooperative (overlaps TDM) ----
        {
            const int j  = tid >> 2;            // 0..63
            const int d0 = (tid & 3) * 32;      // 0,32,64,96
            const bf16_t* vp = V + (size_t)(j0 + j) * CC + h * DD + d0;
            __builtin_prefetch(vp + (size_t)64 * CC, 0, 0);     // next V tile
            __builtin_prefetch(K + (size_t)(j0 + 64 + j) * CC + h * DD + d0, 0, 0);
#pragma unroll
            for (int q = 0; q < 4; ++q) {
                v8bf vv = *(const v8bf*)(vp + 8 * q);
#pragma unroll
                for (int e = 0; e < 8; ++e)
                    Vt[(d0 + 8 * q + e) * 72 + j] = vv[e];
            }
        }
#if USE_TDM
        wait_tensorcnt0();      // no-op for waves that didn't issue
#endif
        __syncthreads();

        // ---- S = Q K^T (16 x 64 per wave) ----
        v8f s[4];
#pragma unroll
        for (int nt = 0; nt < 4; ++nt) {
            v8f sa = v8f_zero();
#pragma unroll
            for (int ks = 0; ks < 4; ++ks) {
#if USE_TDM
                const bf16_t* kp = &Kl[(nt * 16 + ll) * 136 + ks * 32 + 16 * hh];
#else
                const bf16_t* kp = K + (size_t)(j0 + nt * 16 + ll) * CC + h * DD
                                     + ks * 32 + 16 * hh;
#endif
                v16bf kf = load_b_frag(kp);
                sa = __builtin_amdgcn_wmma_f32_16x16x32_bf16(
                        false, qf[ks], false, kf, (short)0, sa, false, false);
            }
            s[nt] = sa;
        }

        // ---- online softmax (rows private to this wave) ----
        float alpha[8];
#pragma unroll
        for (int r = 0; r < 8; ++r) {
            float mx = -1e30f;
#pragma unroll
            for (int nt = 0; nt < 4; ++nt) mx = fmaxf(mx, s[nt][r]);
#pragma unroll
            for (int off = 1; off < 16; off <<= 1)
                mx = fmaxf(mx, __shfl_xor(mx, off, 32));
            mx *= scale;
            const float mnew = fmaxf(mi[r], mx);
            alpha[r] = __expf(mi[r] - mnew);
            float sum = 0.f;
#pragma unroll
            for (int nt = 0; nt < 4; ++nt) {
                const float pv = __expf(s[nt][r] * scale - mnew);
                Pl[(wv * 16 + r + 8 * hh) * 72 + nt * 16 + ll] = f32_to_bf16(pv);
                sum += pv;
            }
#pragma unroll
            for (int off = 1; off < 16; off <<= 1)
                sum += __shfl_xor(sum, off, 32);
            li[r] = li[r] * alpha[r] + sum;
            mi[r] = mnew;
        }
#pragma unroll
        for (int nt = 0; nt < 8; ++nt)
#pragma unroll
            for (int r = 0; r < 8; ++r) o[nt][r] *= alpha[r];

        // ---- O += P V  (P via LDS re-layout, V^T already in LDS) ----
#pragma unroll
        for (int ks = 0; ks < 2; ++ks) {
            v16bf pf = load_a_frag(&Pl[(wv * 16 + ll) * 72 + ks * 32 + 16 * hh]);
#pragma unroll
            for (int nt = 0; nt < 8; ++nt) {
                v16bf vf = load_b_frag(&Vt[(nt * 16 + ll) * 72 + ks * 32 + 16 * hh]);
                o[nt] = __builtin_amdgcn_wmma_f32_16x16x32_bf16(
                            false, pf, false, vf, (short)0, o[nt], false, false);
            }
        }
        __syncthreads();   // before next tile staging
    }

    // ---- epilogue: normalize (reciprocal, not 64 divides) and write fp32 ----
    float inv[8];
#pragma unroll
    for (int r = 0; r < 8; ++r) inv[r] = 1.0f / li[r];
#pragma unroll
    for (int nt = 0; nt < 8; ++nt)
#pragma unroll
        for (int r = 0; r < 8; ++r) {
            const int row = t0 + wv * 16 + r + 8 * hh;
            const int col = h * DD + nt * 16 + ll;
            O[(size_t)row * CC + col] = o[nt][r] * inv[r];
        }
}

// =====================================================================
extern "C" void kernel_launch(void* const* d_in, const int* in_sizes, int n_in,
                              void* d_out, int out_size, void* d_ws, size_t ws_size,
                              hipStream_t stream)
{
    const float* x     = (const float*)d_in[0];
    const float* ctx   = (const float*)d_in[1];
    const float* freqs = (const float*)d_in[2];
    const float* Wq    = (const float*)d_in[3];
    const float* Wk    = (const float*)d_in[4];
    const float* Wv    = (const float*)d_in[5];
    const float* Wo    = (const float*)d_in[6];

    bf16_t* Qb = (bf16_t*)d_ws;
    bf16_t* Kb = Qb + (size_t)TT * CC;
    bf16_t* Vb = Kb + (size_t)TT * CC;
    float*  Af = (float*)(Vb + (size_t)TT * CC);

    const dim3 blk(256);
    const dim3 g_gemm(CC / 128, TT / 128);

    gemm_xw<false><<<g_gemm, blk, 0, stream>>>(x,   Wq, Qb, TT, CC, CC);
    gemm_xw<false><<<g_gemm, blk, 0, stream>>>(ctx, Wk, Kb, TT, CC, CC);
    gemm_xw<false><<<g_gemm, blk, 0, stream>>>(ctx, Wv, Vb, TT, CC, CC);

    const int pairs = TT * HH * (DD / 2);
    rope_kernel<<<(pairs + 255) / 256, blk, 0, stream>>>(Qb, freqs);
    rope_kernel<<<(pairs + 255) / 256, blk, 0, stream>>>(Kb, freqs);

    flash_attn<<<dim3(TT / 128, HH), blk, 0, stream>>>(Qb, Kb, Vb, Af);

    gemm_xw<true><<<g_gemm, blk, 0, stream>>>(Af, Wo, d_out, TT, CC, CC);
}
